// FOLDSNet_3899830305140
// MI455X (gfx1250) — compile-verified, matrix operand order checked by reading, not run on hardware
//
#include <hip/hip_runtime.h>
#include <hip/hip_bf16.h>

typedef __attribute__((ext_vector_type(16))) _Float16 v16h;
typedef __attribute__((ext_vector_type(8)))  float    v8f;

constexpr int kNPix = 3 * 224 * 224;
constexpr int kB    = 512;
constexpr int kRet  = 128;
constexpr int kLgn  = 128;
constexpr int kV1   = 256;
constexpr int kIt   = 128;
constexpr int kCls  = 1000;

__device__ __forceinline__ float sigmoidf(float v) {
    return 1.0f / (1.0f + __expf(-v));
}

// K index for fragment element e, half-wave hi (A and B share this packing).
// lanes 0-15: e 0-7 -> K 0-7,  e 8-15 -> K 16-23
// lanes 16-31: e 0-7 -> K 8-15, e 8-15 -> K 24-31
__device__ __forceinline__ int frag_k(int e, int hi) {
    return ((e >> 3) << 4) + (hi << 3) + (e & 7);
}

// A fragment (16xK tile, row-major source with leading dim ld), f32 -> f16.
__device__ __forceinline__ v16h load_a_frag(const float* src, int ld, int lane, int kbase) {
    const int m  = lane & 15;
    const int hi = (lane >> 4) & 1;
    v16h f;
#pragma unroll
    for (int e = 0; e < 16; ++e)
        f[e] = (_Float16)src[m * ld + kbase + frag_k(e, hi)];
    return f;
}

// B fragment from int 0/1 connectivity, conn[n][k] row-major (N x K), leading dim ld.
__device__ __forceinline__ v16h load_b_frag_conn(const int* conn, int ld, int nbase,
                                                 int lane, int kbase) {
    const int n  = nbase + (lane & 15);
    const int hi = (lane >> 4) & 1;
    v16h f;
#pragma unroll
    for (int e = 0; e < 16; ++e)
        f[e] = (_Float16)(float)conn[n * ld + kbase + frag_k(e, hi)];
    return f;
}

// ---------------------------------------------------------------------------
// Kernel 1: retina -> LGN -> V1 -> IT for a 16-row batch tile (fully fused).
// grid.x = B/16, block = 128 threads (4 waves).
// ---------------------------------------------------------------------------
__global__ __launch_bounds__(128)
void folds_feature_kernel(const float* __restrict__ x,
                          const float* __restrict__ w_ret, const float* __restrict__ b_ret,
                          const float* __restrict__ w_lgn, const float* __restrict__ b_lgn,
                          const float* __restrict__ w_v1,  const float* __restrict__ b_v1,
                          const float* __restrict__ w_it,  const float* __restrict__ b_it,
                          const int*   __restrict__ pixel_map,
                          const int*   __restrict__ lgn_to_v1,
                          const int*   __restrict__ v1_to_it,
                          float*       __restrict__ r4_out) {
    __shared__ float r2s[16 * kLgn];   // (m, l)  8 KB
    __shared__ float r3s[16 * kV1];    // (m, v) 16 KB
    __shared__ float sw_lgn[kLgn];
    __shared__ float sw_v1[kV1];
    __shared__ float sw_it[kIt];

    const int t    = threadIdx.x;
    const int lane = t & 31;
    const int wave = t >> 5;
    const int b0   = blockIdx.x * 16;
    const int hi   = (lane >> 4) & 1;
    const int nl   = lane & 15;

    // ---- Stage 0: per-neuron weight sums (sum over dendritic grid) ----
    if (t < kLgn) {
        float s = 0.f;
        for (int j = 0; j < 16; ++j) s += w_lgn[t * 16 + j];
        sw_lgn[t] = s;
    }
    if (t < kIt) {
        float s = 0.f;
        for (int j = 0; j < 32; ++j) s += w_it[t * 32 + j];
        sw_it[t] = s;
    }
    for (int v = t; v < kV1; v += 128) {
        float s = 0.f;
        for (int j = 0; j < 32; ++j) s += w_v1[v * 32 + j];
        sw_v1[v] = s;
    }

    // ---- Stage 1: retina gather + dot, then LGN elementwise ----
    // idx strides by 128 => every thread keeps n == t (sw_lgn[t] is self-produced).
    for (int idx = t; idx < 16 * kRet; idx += 128) {
        const int n = idx & (kRet - 1);
        const int m = idx >> 7;
        const float* xr = x + (size_t)(b0 + m) * kNPix;
        float acc = 0.f;
#pragma unroll
        for (int j = 0; j < 16; ++j)
            acc += xr[pixel_map[n * 16 + j]] * w_ret[n * 16 + j];
        const float r1 = sigmoidf(acc + b_ret[n]);
        r2s[m * kLgn + n] = sigmoidf(r1 * sw_lgn[n] + b_lgn[n]);
    }
    __syncthreads();

    // ---- Stage 2: V1 = sigmoid( (r2 @ c1^T / deg1) * sw_v1 + b_v1 ) via WMMA ----
    v16h a2[4];
#pragma unroll
    for (int kc = 0; kc < 4; ++kc)
        a2[kc] = load_a_frag(r2s, kLgn, lane, kc * 32);

    for (int tile = wave; tile < kV1 / 16; tile += 4) {
        const int vbase = tile * 16;
        v8f acc = {};
#pragma unroll
        for (int kc = 0; kc < 4; ++kc) {
            v16h bf = load_b_frag_conn(lgn_to_v1, kLgn, vbase, lane, kc * 32);
            acc = __builtin_amdgcn_wmma_f32_16x16x32_f16(false, a2[kc], false, bf,
                                                         (short)0, acc, false, false);
        }
        const int v = vbase + nl;
        int deg = 0;
        for (int l = 0; l < kLgn; ++l) deg += lgn_to_v1[v * kLgn + l];
        const float inv = 1.0f / (float)deg;
        const float sw  = sw_v1[v];
        const float bb  = b_v1[v];
#pragma unroll
        for (int j = 0; j < 8; ++j) {
            const int m = j + (hi << 3);
            r3s[m * kV1 + v] = sigmoidf(acc[j] * inv * sw + bb);
        }
    }
    __syncthreads();

    // ---- Stage 3: IT = sigmoid( (r3 @ c2^T / deg2) * sw_it + b_it ) via WMMA ----
    v16h a3[8];
#pragma unroll
    for (int kc = 0; kc < 8; ++kc)
        a3[kc] = load_a_frag(r3s, kV1, lane, kc * 32);

    for (int tile = wave; tile < kIt / 16; tile += 4) {
        const int ibase = tile * 16;
        v8f acc = {};
#pragma unroll
        for (int kc = 0; kc < 8; ++kc) {
            v16h bf = load_b_frag_conn(v1_to_it, kV1, ibase, lane, kc * 32);
            acc = __builtin_amdgcn_wmma_f32_16x16x32_f16(false, a3[kc], false, bf,
                                                         (short)0, acc, false, false);
        }
        const int i = ibase + nl;
        int deg = 0;
        for (int v = 0; v < kV1; ++v) deg += v1_to_it[i * kV1 + v];
        const float inv = 1.0f / (float)deg;
        const float sw  = sw_it[i];
        const float bb  = b_it[i];
#pragma unroll
        for (int j = 0; j < 8; ++j) {
            const int m = j + (hi << 3);
            r4_out[(size_t)(b0 + m) * kIt + i] = sigmoidf(acc[j] * inv * sw + bb);
        }
    }
}

// ---------------------------------------------------------------------------
// Kernel 2: logits = r4 @ Wc^T + bc, one wave per 16x16 output tile.
// grid = (B/16, ceil(1000/16)), block = 32 threads.
// ---------------------------------------------------------------------------
__global__ __launch_bounds__(32)
void folds_classifier_kernel(const float* __restrict__ r4,
                             const float* __restrict__ Wc,
                             const float* __restrict__ bc,
                             float*       __restrict__ out) {
    const int lane = threadIdx.x & 31;
    const int b0   = blockIdx.x * 16;
    const int n0   = blockIdx.y * 16;
    const int nl   = lane & 15;
    const int hi   = (lane >> 4) & 1;
    const int n    = n0 + nl;
    const bool nvalid = (n < kCls);

    const float* a_src = r4 + (size_t)b0 * kIt;

    v8f acc = {};
#pragma unroll
    for (int kc = 0; kc < 4; ++kc) {
        v16h a = load_a_frag(a_src, kIt, lane, kc * 32);
        v16h b;
#pragma unroll
        for (int e = 0; e < 16; ++e)
            b[e] = nvalid ? (_Float16)Wc[n * kIt + kc * 32 + frag_k(e, hi)]
                          : (_Float16)0.0f;
        acc = __builtin_amdgcn_wmma_f32_16x16x32_f16(false, a, false, b,
                                                     (short)0, acc, false, false);
    }

    if (nvalid) {
        const float bias = bc[n];
#pragma unroll
        for (int j = 0; j < 8; ++j) {
            const int m = j + (hi << 3);
            out[(size_t)(b0 + m) * kCls + n] = acc[j] + bias;
        }
    }
}

extern "C" void kernel_launch(void* const* d_in, const int* in_sizes, int n_in,
                              void* d_out, int out_size, void* d_ws, size_t ws_size,
                              hipStream_t stream) {
    const float* x     = (const float*)d_in[0];
    const float* w_ret = (const float*)d_in[1];
    const float* b_ret = (const float*)d_in[2];
    const float* w_lgn = (const float*)d_in[3];
    const float* b_lgn = (const float*)d_in[4];
    const float* w_v1  = (const float*)d_in[5];
    const float* b_v1  = (const float*)d_in[6];
    const float* w_it  = (const float*)d_in[7];
    const float* b_it  = (const float*)d_in[8];
    const float* Wc    = (const float*)d_in[9];
    const float* bc    = (const float*)d_in[10];
    const int* pixel_map = (const int*)d_in[11];
    const int* lgn_to_v1 = (const int*)d_in[12];
    const int* v1_to_it  = (const int*)d_in[13];

    float* out = (float*)d_out;
    float* r4  = (float*)d_ws;   // B x 128 f32 = 256 KB scratch

    folds_feature_kernel<<<dim3(kB / 16), dim3(128), 0, stream>>>(
        x, w_ret, b_ret, w_lgn, b_lgn, w_v1, b_v1, w_it, b_it,
        pixel_map, lgn_to_v1, v1_to_it, r4);

    folds_classifier_kernel<<<dim3(kB / 16, (kCls + 15) / 16), dim3(32), 0, stream>>>(
        r4, Wc, bc, out);
}
